// MCGraphAttention_59373627900284
// MI455X (gfx1250) — compile-verified, matrix operand order checked by reading, not run on hardware
//
#include <hip/hip_runtime.h>
#include <hip/hip_bf16.h>
#include <stdint.h>

typedef __attribute__((ext_vector_type(16))) _Float16 v16h;
typedef __attribute__((ext_vector_type(8)))  _Float16 v8h;
typedef __attribute__((ext_vector_type(8)))  float    v8f;

#define NN   4096
#define EMB  256
#define NH   4
#define HC   64
#define NEG_SLOPE 0.2f
#define NEG_BIG  -3.0e38f

static __device__ __forceinline__ v8f wmma_f16(v16h a, v16h b, v8f c) {
  // D = A(16x32 f16) * B(32x16 f16) + C(16x16 f32)
  return __builtin_amdgcn_wmma_f32_16x16x32_f16(false, a, false, b, (short)0, c,
                                                false, false);
}

// LDS byte offset of a generic pointer to __shared__ (low 32 bits are the
// wave-relative LDS address per CDNA5 flat/async LDS addressing).
static __device__ __forceinline__ unsigned lds_off(const void* p) {
  return (unsigned)(uintptr_t)p;
}

// ------------------- pack W (f32 row-major) into WMMA B-fragment layout (f16)
// packed index i = ((ntile*8 + kb)*32 + lane)*16 + t
//   kchunk(t,lane) = t<8 ? koff+t : 16+koff+(t-8),  koff = (lane>=16)*8
__global__ __launch_bounds__(256) void k_pack(const float* __restrict__ a,
                                              const float* __restrict__ b,
                                              _Float16* __restrict__ oa,
                                              _Float16* __restrict__ ob) {
  const int i  = blockIdx.x * 256 + threadIdx.x;   // 0..65535
  const int t  = i & 15;
  const int l  = (i >> 4) & 31;
  const int kb = (i >> 9) & 7;
  const int nt = i >> 12;
  const int koff = ((l >> 4) & 1) * 8;
  const int kk  = kb * 32 + ((t < 8) ? (koff + t) : (16 + koff + (t - 8)));
  const int col = nt * 16 + (l & 15);
  oa[i] = (_Float16)a[kk * EMB + col];
  ob[i] = (_Float16)b[kk * EMB + col];
}

// --------------------- h = x @ W_lin (WMMA f16); writes h row-major f32 for
// the coefficient kernel AND fragment-major f16 (hp16) for the GAT kernel.
// hp16 layout: hp16[((jblk*16 + ct)*32 + lane)*16 + t] = h[jblk*32 + jchunk(t,lane)][ct*16 + (lane&15)]
__global__ __launch_bounds__(32) void k_gemm_h(const float* __restrict__ x,
                                               const _Float16* __restrict__ wp16,
                                               float* __restrict__ h32,
                                               _Float16* __restrict__ hp16) {
  const int i0   = blockIdx.x * 16;
  const int n0   = blockIdx.y * 16;
  const int l    = threadIdx.x;          // lane 0..31
  const int rsel = (l >> 4) & 1;
  const int koff = rsel * 8;
  const int row  = i0 + (l & 15);
  v8f acc = {};
  const float* xr = x + (size_t)row * EMB;
  const _Float16* wp = wp16 + (size_t)(n0 >> 4) * 4096 + (size_t)l * 16;
#pragma unroll
  for (int kb = 0; kb < EMB / 32; ++kb) {
    const int k = kb * 32;
    v16h a, b;
    const v8h blo = *(const v8h*)(wp + (size_t)kb * 512);
    const v8h bhi = *(const v8h*)(wp + (size_t)kb * 512 + 8);
#pragma unroll
    for (int t = 0; t < 8; ++t) {
      a[t]     = (_Float16)xr[k + koff + t];
      a[8 + t] = (_Float16)xr[k + 16 + koff + t];
      b[t]     = blo[t];
      b[8 + t] = bhi[t];
    }
    acc = wmma_f16(a, b, acc);
  }
  const int rbase = i0 + rsel * 8;
  const int col   = n0 + (l & 15);
#pragma unroll
  for (int v = 0; v < 8; ++v) {
    const int j = rbase + v;
    h32[(size_t)j * EMB + col] = acc[v];
    // inverse fragment permutation: jloc -> (rsel', t')
    const int jloc = j & 31;
    const int rs2  = (jloc >> 3) & 1;
    const int t2   = (jloc & 7) + ((jloc >> 4) & 1) * 8;
    const size_t idx =
        ((size_t)((j >> 5) * 16 + (col >> 4)) * 32 + rs2 * 16 + (col & 15)) * 16 + t2;
    hp16[idx] = (_Float16)acc[v];
  }
}

// ---------------- a_src, a_dst dot products, stored TRANSPOSED: [H][N]
__global__ __launch_bounds__(256) void k_coef(const float* __restrict__ h32,
                                              const float* __restrict__ att_src,
                                              const float* __restrict__ att_dst,
                                              float* __restrict__ asrc_t,
                                              float* __restrict__ adst_t) {
  const int idx = blockIdx.x * 256 + threadIdx.x;   // NN*NH
  const int n = idx >> 2, hd = idx & 3;
  const float* hp = h32 + (size_t)n * EMB + hd * HC;
  float ss = 0.f, sd = 0.f;
#pragma unroll 4
  for (int c = 0; c < HC; ++c) {
    const float hv = hp[c];
    ss += hv * att_src[hd * HC + c];
    sd += hv * att_dst[hd * HC + c];
  }
  asrc_t[(size_t)hd * NN + n] = ss;
  adst_t[(size_t)hd * NN + n] = sd;
}

// --- issue the 10 async global->LDS ops for one j-block into one LDS buffer
static __device__ __forceinline__ void stage_async(const _Float16* hp16,
                                                   const float* asrc_t,
                                                   const int* adj,
                                                   unsigned lb_h, unsigned lb_adj,
                                                   unsigned lb_as,
                                                   int i0, int jb, int tid) {
  const char* gsrc = (const char*)(hp16 + (size_t)jb * 8192);
#pragma unroll
  for (int t = 0; t < 8; ++t) {   // 16 KB fragment-major h j-block
    const unsigned dst = lb_h + (unsigned)((tid + t * 128) * 16);
    const unsigned long long src =
        (unsigned long long)(uintptr_t)(gsrc + (size_t)(tid + t * 128) * 16);
    asm volatile("global_load_async_to_lds_b128 %0, %1, off"
                 :: "v"(dst), "v"(src) : "memory");
  }
  // adjacency tile [16 x 32] ints, one b128 per thread
  const int rr = tid >> 3, cq = tid & 7;
  const unsigned dsta = lb_adj + (unsigned)(tid * 16);
  const unsigned long long srca =
      (unsigned long long)(uintptr_t)(adj + (size_t)(i0 + rr) * NN + jb * 32 + cq * 4);
  asm volatile("global_load_async_to_lds_b128 %0, %1, off"
               :: "v"(dsta), "v"(srca) : "memory");
  // a_src slices (transposed layout -> contiguous per head), one b32 per thread
  const int hd2 = tid >> 5, jj = tid & 31;
  const unsigned dsts = lb_as + (unsigned)(tid * 4);
  const unsigned long long srcs =
      (unsigned long long)(uintptr_t)(asrc_t + (size_t)hd2 * NN + jb * 32 + jj);
  asm volatile("global_load_async_to_lds_b32 %0, %1, off"
               :: "v"(dsts), "v"(srcs) : "memory");
}

// ------------- flash-style masked-softmax aggregation: out = softmax(alpha)@h
// block = 128 threads = 4 waves (one head each), 16 dest rows per block.
// Double-buffered CDNA5 async global->LDS DMA overlaps staging with compute:
// group of 10 async ops per block; s_wait_asynccnt 0xa retires the previous
// group (async loads complete in order) while the next stays in flight.
__global__ __launch_bounds__(128) void k_gat(const _Float16* __restrict__ hp16,
                                             const float* __restrict__ asrc_t,
                                             const float* __restrict__ adst_t,
                                             const int* __restrict__ adj,
                                             const float* __restrict__ bias_att,
                                             _Float16* __restrict__ og16) {
  __shared__ __align__(16) _Float16 lds_h[2][16 * 512];  // 2 x 16 KB frag-major
  __shared__ __align__(16) int      lds_adj[2][16 * 32]; // 2 x 2 KB
  __shared__ __align__(16) float    lds_as[2][NH * 32];  // 2 x 512 B
  __shared__ float                  lds_bc[NH][16];      // per-row broadcast

  const int i0   = blockIdx.x * 16;
  const int tid  = threadIdx.x;
  const int hd   = tid >> 5;             // head = wave id
  const int l    = tid & 31;
  const int r    = l & 15;               // dest row within tile
  const int rsel = (l >> 4) & 1;
  const int koff = rsel * 8;

  const float adst_r = adst_t[(size_t)hd * NN + i0 + r];

  const unsigned lb_h[2]   = {lds_off(&lds_h[0][0]),   lds_off(&lds_h[1][0])};
  const unsigned lb_adj[2] = {lds_off(&lds_adj[0][0]), lds_off(&lds_adj[1][0])};
  const unsigned lb_as[2]  = {lds_off(&lds_as[0][0]),  lds_off(&lds_as[1][0])};

  v8f acc[4] = {{}, {}, {}, {}};         // 16 x 64 f32 per head
  float mrow = NEG_BIG, lrow = 0.0f;

  const int NB = NN / 32;
  stage_async(hp16, asrc_t, adj, lb_h[0], lb_adj[0], lb_as[0], i0, 0, tid);

  for (int jb = 0; jb < NB; ++jb) {
    const int bf = jb & 1;
    const int j0 = jb * 32;
    __syncthreads();                     // readers of buffer bf^1 are done
    if (jb + 1 < NB) {
      stage_async(hp16, asrc_t, adj, lb_h[bf ^ 1], lb_adj[bf ^ 1],
                  lb_as[bf ^ 1], i0, jb + 1, tid);
      asm volatile("s_wait_asynccnt 0xa" ::: "memory");  // retire group for bf
    } else {
      asm volatile("s_wait_asynccnt 0x0" ::: "memory");
    }
    __syncthreads();                     // publish buffer bf to all waves

    // vector DS reads of a_src and adj in A-fragment order
    float fas[16];
    int   madj[16];
    {
      const float* ap = &lds_as[bf][hd * 32];
      *(float4*)&fas[0]  = *(const float4*)&ap[koff];
      *(float4*)&fas[4]  = *(const float4*)&ap[koff + 4];
      *(float4*)&fas[8]  = *(const float4*)&ap[16 + koff];
      *(float4*)&fas[12] = *(const float4*)&ap[16 + koff + 4];
      const int* jp = &lds_adj[bf][r * 32];
      *(int4*)&madj[0]   = *(const int4*)&jp[koff];
      *(int4*)&madj[4]   = *(const int4*)&jp[koff + 4];
      *(int4*)&madj[8]   = *(const int4*)&jp[16 + koff];
      *(int4*)&madj[12]  = *(const int4*)&jp[16 + koff + 4];
    }

    float s[16];
    float pmax = NEG_BIG;
#pragma unroll
    for (int t = 0; t < 16; ++t) {
      const int jloc = (t < 8) ? (koff + t) : (16 + koff + (t - 8));
      float sv = fas[t] + adst_r;
      sv = (sv >= 0.0f) ? sv : NEG_SLOPE * sv;        // leaky relu
      const bool m = (madj[t] != 0) | ((i0 + r) == (j0 + jloc));
      s[t] = m ? sv : NEG_BIG;
      pmax = fmaxf(pmax, s[t]);
    }
    pmax = fmaxf(pmax, __shfl_xor(pmax, 16, 32));     // join lane halves
    const float mnew  = fmaxf(mrow, pmax);
    const float scale = (mrow <= -1e37f) ? 0.0f : __expf(mrow - mnew);
    float psum = 0.0f;
    v16h aP;
#pragma unroll
    for (int t = 0; t < 16; ++t) {
      const float p = (s[t] <= -1e37f) ? 0.0f : __expf(s[t] - mnew);
      psum += p;
      aP[t] = (_Float16)p;
    }
    psum += __shfl_xor(psum, 16, 32);
    lrow = lrow * scale + psum;
    mrow = mnew;

    if (l < 16) lds_bc[hd][l] = scale;   // broadcast row-rescale to D layout
    __syncthreads();
    const int rbase = rsel * 8;
    float rs[8];
#pragma unroll
    for (int v = 0; v < 8; ++v) rs[v] = lds_bc[hd][rbase + v];
#pragma unroll
    for (int cc = 0; cc < 4; ++cc)
#pragma unroll
      for (int v = 0; v < 8; ++v) acc[cc][v] *= rs[v];

    // P(16x32) @ h_block(32x64): B operands are contiguous fragment reads
#pragma unroll
    for (int cc = 0; cc < 4; ++cc) {
      const _Float16* hb = &lds_h[bf][(size_t)(hd * 4 + cc) * 512 + l * 16];
      const v8h lo = *(const v8h*)hb;
      const v8h hi = *(const v8h*)(hb + 8);
      v16h b;
#pragma unroll
      for (int t = 0; t < 8; ++t) { b[t] = lo[t]; b[8 + t] = hi[t]; }
      acc[cc] = wmma_f16(aP, b, acc[cc]);
    }
  }

  __syncthreads();
  if (l < 16) lds_bc[hd][l] = lrow;      // broadcast row sums
  __syncthreads();
  const int rbase = rsel * 8;
#pragma unroll
  for (int cc = 0; cc < 4; ++cc) {
    const int col = hd * HC + cc * 16 + (l & 15);
    const float ba = bias_att[col];
#pragma unroll
    for (int v = 0; v < 8; ++v) {
      const float inv = 1.0f / lds_bc[hd][rbase + v];
      og16[(size_t)(i0 + rbase + v) * EMB + col] = (_Float16)(acc[cc][v] * inv + ba);
    }
  }
}

// --------------------- y = out @ W_out + b_out, then LayerNorm (fused)
__global__ __launch_bounds__(512) void k_outln(const _Float16* __restrict__ og16,
                                               const _Float16* __restrict__ wp16,
                                               const float* __restrict__ b_out,
                                               const float* __restrict__ gamma,
                                               const float* __restrict__ beta,
                                               float* __restrict__ out) {
  __shared__ float lds_y[16 * EMB];
  const int i0   = blockIdx.x * 16;
  const int w    = threadIdx.x >> 5;     // wave id = col tile (16 waves)
  const int l    = threadIdx.x & 31;
  const int rsel = (l >> 4) & 1;
  const int koff = rsel * 8;
  const int n0   = w * 16;
  const int row  = i0 + (l & 15);
  const int col  = n0 + (l & 15);
  const _Float16* wp = wp16 + (size_t)w * 4096 + (size_t)l * 16;
  v8f acc = {};
#pragma unroll
  for (int kb = 0; kb < EMB / 32; ++kb) {
    const int k = kb * 32;
    v16h a, b;
    const v8h alo = *(const v8h*)(og16 + (size_t)row * EMB + k + koff);
    const v8h ahi = *(const v8h*)(og16 + (size_t)row * EMB + k + 16 + koff);
    const v8h blo = *(const v8h*)(wp + (size_t)kb * 512);
    const v8h bhi = *(const v8h*)(wp + (size_t)kb * 512 + 8);
#pragma unroll
    for (int t = 0; t < 8; ++t) {
      a[t]     = alo[t];
      a[8 + t] = ahi[t];
      b[t]     = blo[t];
      b[8 + t] = bhi[t];
    }
    acc = wmma_f16(a, b, acc);
  }
  const int rbase = rsel * 8;
  const float bo = b_out[col];
#pragma unroll
  for (int v = 0; v < 8; ++v) lds_y[(rbase + v) * EMB + col] = acc[v] + bo;
  __syncthreads();

  // LayerNorm: wave w normalizes row w (256 cols, 8 per lane)
  float vals[8], sum = 0.f, sq = 0.f;
#pragma unroll
  for (int q = 0; q < 8; ++q) {
    const float v = lds_y[w * EMB + l + 32 * q];
    vals[q] = v; sum += v; sq += v * v;
  }
#pragma unroll
  for (int off = 16; off > 0; off >>= 1) {
    sum += __shfl_xor(sum, off, 32);
    sq  += __shfl_xor(sq,  off, 32);
  }
  const float mu  = sum * (1.0f / EMB);
  const float var = sq * (1.0f / EMB) - mu * mu;
  const float inv = rsqrtf(var + 1e-5f);
#pragma unroll
  for (int q = 0; q < 8; ++q) {
    const int c = l + 32 * q;
    out[(size_t)(i0 + w) * EMB + c] = (vals[q] - mu) * inv * gamma[c] + beta[c];
  }
}

extern "C" void kernel_launch(void* const* d_in, const int* in_sizes, int n_in,
                              void* d_out, int out_size, void* d_ws, size_t ws_size,
                              hipStream_t stream) {
  const float* x       = (const float*)d_in[0];
  const int*   adj     = (const int*)  d_in[1];
  const float* W_lin   = (const float*)d_in[2];
  const float* att_src = (const float*)d_in[3];
  const float* att_dst = (const float*)d_in[4];
  const float* biasA   = (const float*)d_in[5];
  const float* W_out   = (const float*)d_in[6];
  const float* b_out   = (const float*)d_in[7];
  const float* gamma   = (const float*)d_in[8];
  const float* beta    = (const float*)d_in[9];
  float* out = (float*)d_out;

  char* ws = (char*)d_ws;
  float*    h32  = (float*)   (ws + 0);        // 4096*256 f32  = 4 MB
  _Float16* hp16 = (_Float16*)(ws + 4194304);  // frag-major h, f16 = 2 MB
  float*    asrc = (float*)   (ws + 6291456);  // [H][N] f32
  float*    adst = (float*)   (ws + 6356992);  // [H][N] f32
  _Float16* wl16 = (_Float16*)(ws + 6422528);  // packed 256*256 f16
  _Float16* wo16 = (_Float16*)(ws + 6553600);  // packed 256*256 f16
  _Float16* og16 = (_Float16*)(ws + 6684672);  // 4096*256 f16 = 2 MB

  hipLaunchKernelGGL(k_pack,   dim3(256),            dim3(256), 0, stream,
                     W_lin, W_out, wl16, wo16);
  hipLaunchKernelGGL(k_gemm_h, dim3(NN/16, EMB/16),  dim3(32),  0, stream,
                     x, wl16, h32, hp16);
  hipLaunchKernelGGL(k_coef,   dim3(NN*NH/256),      dim3(256), 0, stream,
                     h32, att_src, att_dst, asrc, adst);
  hipLaunchKernelGGL(k_gat,    dim3(NN/16),          dim3(128), 0, stream,
                     hp16, asrc, adst, adj, biasA, og16);
  hipLaunchKernelGGL(k_outln,  dim3(NN/16),          dim3(512), 0, stream,
                     og16, wo16, b_out, gamma, beta, out);
}
